// Lista_46059229282426
// MI455X (gfx1250) — compile-verified, matrix operand order checked by reading, not run on hardware
//
#include <hip/hip_runtime.h>
#include <stdint.h>

typedef __attribute__((ext_vector_type(16))) _Float16 v16h;
typedef __attribute__((ext_vector_type(8)))  float    v8f;

#define WMMA_F32_F16(A, B, C) \
  __builtin_amdgcn_wmma_f32_16x16x32_f16(false, (A), false, (B), (short)0, (C), false, false)

// pack two f32 into one dword of two f16 (v_cvt_pk_rtz_f16_f32)
__device__ __forceinline__ uint32_t pkrtz_u32(float a, float b) {
  auto h = __builtin_amdgcn_cvt_pkrtz(a, b);
  union { decltype(h) h2; uint32_t u; } cv;
  cv.h2 = h;
  return cv.u;
}

// Two stacked 16x16 f32 D tiles (K = 0..15 in d0, K = 16..31 in d1) -> f16 B operand.
// Lane-local: B vgpr v holds K = (v>>2)*16 + half*8 + (v&3)*2 (+1 packed), which for the
// same lane corresponds to D regs 2v,2v+1 of the matching tile.
__device__ __forceinline__ v16h packB(const v8f& d0, const v8f& d1) {
  union { uint32_t u[8]; v16h h; } p;
#pragma unroll
  for (int j = 0; j < 4; ++j) {
    p.u[j]     = pkrtz_u32(d0[2 * j], d0[2 * j + 1]);
    p.u[4 + j] = pkrtz_u32(d1[2 * j], d1[2 * j + 1]);
  }
  return p.h;
}

// Load one 16x32 f16 A tile from LDS (stored as [tile][lane][8] dwords).
__device__ __forceinline__ v16h loadA(const uint32_t* __restrict__ base, int lane) {
  union { uint32_t u[8]; v16h h; } t;
  const uint32_t* p = base + lane * 8;
#pragma unroll
  for (int i = 0; i < 8; ++i) t.u[i] = p[i];
  return t.h;
}

// Problem constants (from setup_inputs): B=4, H=W=192, K=9 (81 taps), F=64, Ho=Wo=184,
// unfoldings=24 -> 23 refinement iterations.
__global__ __launch_bounds__(256) void lista_main(
    const float* __restrict__ I, const float* __restrict__ WA,
    const float* __restrict__ WD, const float* __restrict__ WWm,
    const float* __restrict__ lmb, float* __restrict__ acc) {
  __shared__ uint32_t lds[36 * 32 * 8];  // 36 A-tiles (12 WA, 12 WD, 12 WW), 1KB each
  const int tid = threadIdx.x;

  // ---- stage all weights into LDS in exact WMMA A-operand layout (f16, zero-padded) ----
  {
    const int lane8 = tid >> 3;  // A-operand lane 0..31
    const int v     = tid & 7;   // A-operand vgpr 0..7
    const int m15   = lane8 & 15;
    const int hf8   = lane8 >> 4;
    const int kOff  = ((v >> 2) << 4) + (hf8 << 3) + ((v & 3) << 1);
#pragma unroll 1
    for (int t = 0; t < 36; ++t) {
      float f0 = 0.f, f1 = 0.f;
      if (t < 12) {  // WA: [64 x 81] -> tiles [4 Mtiles x 3 Ksteps], K padded to 96
        int mt = t / 3, s = t - mt * 3;
        int row = mt * 16 + m15;
        int K0  = s * 32 + kOff;
        if (K0 < 81)     f0 = WA[row * 81 + K0];
        if (K0 + 1 < 81) f1 = WA[row * 81 + K0 + 1];
      } else {       // WD / WW: [81 x 64] -> tiles [6 Mtiles(pad 96) x 2 Ksteps]
        int u = t - 12;
        const float* Wp = WD;
        if (u >= 12) { u -= 12; Wp = WWm; }
        int xt = u >> 1, ks = u & 1;
        int row = xt * 16 + m15;
        int K0  = ks * 32 + kOff;
        if (row < 81) { f0 = Wp[row * 64 + K0]; f1 = Wp[row * 64 + K0 + 1]; }
      }
      lds[(t * 32 + lane8) * 8 + v] = pkrtz_u32(f0, f1);
    }
  }
  __syncthreads();

  const uint32_t* A_wa = lds;
  const uint32_t* A_wd = lds + 12 * 256;
  const uint32_t* A_ww = lds + 24 * 256;

  const int lane = tid & 31;
  const int col  = lane & 15;   // position within the wave's N=16 tile
  const int hf   = lane >> 4;   // which half of the D rows this lane holds
  const int wv   = tid >> 5;

  const int pos = (blockIdx.x * 8 + wv) * 16 + col;  // grid sized exactly: no OOB
  const int b   = pos / (184 * 184);
  const int rem = pos - b * (184 * 184);
  const int oy  = rem / 184;
  const int ox  = rem - oy * 184;
  const float* Ib = I + ((size_t)(b * 192 + oy)) * 192 + ox;

  // per-filter threshold, D-layout aligned
  float lam[4][8];
#pragma unroll
  for (int mt = 0; mt < 4; ++mt)
#pragma unroll
    for (int r = 0; r < 8; ++r) lam[mt][r] = lmb[mt * 16 + hf * 8 + r];

  // ---- gather patch, compute per-patch mean, build centered I_col as f16 B operands ----
  float va[3][8], vb[3][8];
  float sum = 0.f;
#pragma unroll
  for (int s = 0; s < 3; ++s)
#pragma unroll
    for (int v = 0; v < 8; ++v) {
      const int c0 = s * 32 + ((v >> 2) << 4) + (hf << 3) + ((v & 3) << 1);
      const int c1 = c0 + 1;
      float f0 = 0.f, f1 = 0.f;
      if (c0 < 81) f0 = Ib[(c0 / 9) * 192 + (c0 % 9)];
      if (c1 < 81) f1 = Ib[(c1 / 9) * 192 + (c1 % 9)];
      va[s][v] = f0; vb[s][v] = f1;
      sum += f0 + f1;
    }
  sum += __shfl_xor(sum, 16, 32);  // combine the two half-wave partials of each column
  const float mean = sum * (1.0f / 81.0f);

  v16h IcB[3];
#pragma unroll
  for (int s = 0; s < 3; ++s) {
    union { uint32_t u[8]; v16h h; } p;
#pragma unroll
    for (int v = 0; v < 8; ++v) {
      const int c0 = s * 32 + ((v >> 2) << 4) + (hf << 3) + ((v & 3) << 1);
      float f0 = (c0 < 81)     ? va[s][v] - mean : 0.f;
      float f1 = (c0 + 1 < 81) ? vb[s][v] - mean : 0.f;
      p.u[v] = pkrtz_u32(f0, f1);
    }
    IcB[s] = p.h;
  }

  // ---- e0 = WA * I_col ; gamma0 = softthr(e0) ; keep cenc = -e0 as encode C operand ----
  v8f zeroC = {};
  v8f cenc[4], gam[4];
#pragma unroll
  for (int mt = 0; mt < 4; ++mt) {
    v8f a = zeroC;
#pragma unroll
    for (int s = 0; s < 3; ++s)
      a = WMMA_F32_F16(loadA(A_wa + (mt * 3 + s) * 256, lane), IcB[s], a);
    cenc[mt] = -a;
#pragma unroll
    for (int r = 0; r < 8; ++r) {
      float u = a[r];
      float t = fmaxf(fabsf(u) - lam[mt][r], 0.f);
      gam[mt][r] = copysignf(t, u);
    }
  }

  // ---- 23 LISTA unfoldings, gamma fully register-resident ----
#pragma unroll 1
  for (int it = 0; it < 23; ++it) {
    v16h gB0 = packB(gam[0], gam[1]);  // filters 0..31 as B operand
    v16h gB1 = packB(gam[2], gam[3]);  // filters 32..63
    v8f racc[4];
#pragma unroll
    for (int s = 0; s < 3; ++s) {
      // decode x rows [32s, 32s+32): two M tiles of WD, K over 64 filters
      v8f x0 = WMMA_F32_F16(loadA(A_wd + ((2 * s) * 2 + 0) * 256, lane), gB0, zeroC);
      x0     = WMMA_F32_F16(loadA(A_wd + ((2 * s) * 2 + 1) * 256, lane), gB1, x0);
      v8f x1 = WMMA_F32_F16(loadA(A_wd + ((2 * s + 1) * 2 + 0) * 256, lane), gB0, zeroC);
      x1     = WMMA_F32_F16(loadA(A_wd + ((2 * s + 1) * 2 + 1) * 256, lane), gB1, x1);
      v16h xB = packB(x0, x1);  // lane-local D->B repack
      // encode partial: racc = WA_s * x_s + (s==0 ? -e0 : racc)  => racc = WA*x - WA*I_col
#pragma unroll
      for (int mt = 0; mt < 4; ++mt)
        racc[mt] = WMMA_F32_F16(loadA(A_wa + (mt * 3 + s) * 256, lane), xB,
                                (s == 0) ? cenc[mt] : racc[mt]);
    }
#pragma unroll
    for (int mt = 0; mt < 4; ++mt)
#pragma unroll
      for (int r = 0; r < 8; ++r) {
        float u = gam[mt][r] - racc[mt][r];
        float t = fmaxf(fabsf(u) - lam[mt][r], 0.f);
        gam[mt][r] = copysignf(t, u);
      }
  }

  // ---- out_all = WW * gamma + mean ; overlap-add scatter (col2im numerator) ----
  v16h gB0 = packB(gam[0], gam[1]);
  v16h gB1 = packB(gam[2], gam[3]);
#pragma unroll
  for (int xt = 0; xt < 6; ++xt) {
    v8f o = WMMA_F32_F16(loadA(A_ww + (xt * 2 + 0) * 256, lane), gB0, zeroC);
    o     = WMMA_F32_F16(loadA(A_ww + (xt * 2 + 1) * 256, lane), gB1, o);
#pragma unroll
    for (int r = 0; r < 8; ++r) {
      const int c = xt * 16 + hf * 8 + r;
      if (c < 81) {
        const int kh = c / 9, kw = c - kh * 9;
        atomicAdd(acc + ((size_t)(b * 192 + oy + kh)) * 192 + (ox + kw), o[r] + mean);
      }
    }
  }
}

__global__ void lista_zero(float* __restrict__ acc, int n) {
  int i = blockIdx.x * blockDim.x + threadIdx.x;
  if (i < n) acc[i] = 0.f;
}

// divide by analytic overlap count: cnt(y,x) = cy(y)*cx(x)
__global__ void lista_finalize(const float* __restrict__ acc, float* __restrict__ out, int n) {
  int i = blockIdx.x * blockDim.x + threadIdx.x;
  if (i >= n) return;
  int x = i % 192;
  int y = (i / 192) % 192;
  int cy = (y < 8 ? y : 8) - (y > 183 ? y - 183 : 0) + 1;
  int cx = (x < 8 ? x : 8) - (x > 183 ? x - 183 : 0) + 1;
  out[i] = acc[i] / (float)(cy * cx);
}

extern "C" void kernel_launch(void* const* d_in, const int* in_sizes, int n_in,
                              void* d_out, int out_size, void* d_ws, size_t ws_size,
                              hipStream_t stream) {
  (void)in_sizes; (void)n_in; (void)out_size; (void)ws_size;
  const float* I   = (const float*)d_in[0];  // [4,1,192,192]
  const float* WA  = (const float*)d_in[1];  // [64,81]
  const float* WD  = (const float*)d_in[2];  // [81,64]
  const float* WWm = (const float*)d_in[3];  // [81,64]
  const float* lmb = (const float*)d_in[4];  // [64]
  float* out = (float*)d_out;
  float* acc = (float*)d_ws;  // f32 accumulator, 4*192*192 elements

  const int npix = 4 * 192 * 192;
  lista_zero<<<(npix + 255) / 256, 256, 0, stream>>>(acc, npix);
  // 135,424 positions, 16 per wave, 8 waves per block -> exactly 1058 blocks
  lista_main<<<(4 * 184 * 184) / 128, 256, 0, stream>>>(I, WA, WD, WWm, lmb, acc);
  lista_finalize<<<(npix + 255) / 256, 256, 0, stream>>>(acc, out, npix);
}
